// InvariantPointAttention_23905787969873
// MI455X (gfx1250) — matrix operand order repaired and checked by reading, not compiled, not run on hardware
//
#include <hip/hip_runtime.h>
#include <hip/hip_bf16.h>

// ---------------- problem constants ----------------
static constexpr int Bb  = 2;
static constexpr int Nn  = 512;
static constexpr int CSc = 256;
static constexpr int CZc = 128;
static constexpr int Cc  = 128;
static constexpr int Hh  = 8;
static constexpr int PQKp = 8;
static constexpr int PVv  = 12;
static constexpr int FEAT = Hh*CZc + Hh*Cc + Hh*PVv*4;   // 2432
static constexpr int HC   = Hh*Cc;                       // 1024
static constexpr int RQ   = HC;                          // wq rows
static constexpr int RKV  = 2*HC;                        // wkv rows
static constexpr int RQP  = Hh*PQKp*3;                   // 192
static constexpr int RKVP = Hh*(PQKp+PVv)*3;             // 480

static constexpr float SC_QK  = 0.05103103630798287f;    // sqrt(1/(3*128))
static constexpr float SC_B   = 0.5773502691896258f;     // sqrt(1/3)
static constexpr float SC_PT  = 0.09622504486493764f;    // sqrt(1/(3*(PQK*9/2)))
static constexpr float LN1E4  = 9.210340371976184f;      // ln(10000)

typedef __attribute__((ext_vector_type(16))) _Float16 v16h;
typedef __attribute__((ext_vector_type(8)))  float    v8f;
typedef __attribute__((ext_vector_type(4)))  unsigned int u32x4;
typedef __attribute__((ext_vector_type(8)))  int          i32x8;
typedef __attribute__((ext_vector_type(4)))  int          i32x4;

#define DEVI __device__ __forceinline__

// ---- gfx1250 data-mover feature detection (device pass only) ----
#if defined(__AMDGCN__) && __has_builtin(__builtin_amdgcn_tensor_load_to_lds)
#  define USE_TDM 1
#  if __has_include(<hip/amd_detail/amd_gfx1250_TDM.h>)
#    define TDM_ARGS6 1      // therock-10.0 headers -> 6-arg builtin
#  else
#    define TDM_ARGS6 0      // ROCm 7.2 -> 5-arg builtin
#  endif
#elif defined(__AMDGCN__) && __has_builtin(__builtin_amdgcn_global_load_async_to_lds_b128)
#  define USE_ASYNC 1
#endif

// Emit the selected staging path into the compile log (device pass only).
#if defined(__AMDGCN__)
#  if defined(USE_TDM) && TDM_ARGS6
#    pragma message("ATHENA-PATH: TDM tensor_load_to_lds (6-arg)")
#  elif defined(USE_TDM)
#    pragma message("ATHENA-PATH: TDM tensor_load_to_lds (5-arg)")
#  elif defined(USE_ASYNC)
#    pragma message("ATHENA-PATH: global_load_async_to_lds_b128")
#  else
#    pragma message("ATHENA-PATH: synchronous LDS staging fallback")
#  endif
#endif

DEVI void wait_tensor0() {
#if __has_builtin(__builtin_amdgcn_s_wait_tensorcnt)
  __builtin_amdgcn_s_wait_tensorcnt(0);
#elif defined(__AMDGCN__)
  asm volatile("s_wait_tensorcnt 0x0" ::: "memory");
#endif
}

DEVI void wait_async0() {
#if __has_builtin(__builtin_amdgcn_s_wait_asynccnt)
  __builtin_amdgcn_s_wait_asynccnt(0);
#elif defined(__AMDGCN__)
  asm volatile("s_wait_asynccnt 0x0" ::: "memory");
#endif
}

#if defined(USE_TDM)
// Issue one TDM 2D tile load: 32 rows x 128 f16, row stride 128 elements.
// lds_byte = LDS byte address of destination, gaddr = global byte address.
DEVI void tdm_load_tile_32x128h(unsigned lds_byte, unsigned long long gaddr) {
  u32x4 g0;
  g0[0] = 1u;                                            // count=1, user desc
  g0[1] = (unsigned)__builtin_amdgcn_readfirstlane((int)lds_byte);
  g0[2] = (unsigned)__builtin_amdgcn_readfirstlane((int)(unsigned)gaddr);
  g0[3] = ((unsigned)__builtin_amdgcn_readfirstlane((int)(unsigned)(gaddr >> 32))
           & 0x01FFFFFFu) | 0x80000000u;                 // type=2 ("image")
  i32x8 g1;
  g1[0] = 0x00010000;            // workgroup_mask=0, data_size=1 (2 bytes)
  g1[1] = (int)(128u << 16);     // tensor_dim0[15:0] = 128
  g1[2] = (int)(512u << 16);     // tensor_dim0[31:16]=0 | tensor_dim1[15:0]=512
  g1[3] = (int)(128u << 16);     // tensor_dim1[31:16]=0 | tile_dim0 = 128
  g1[4] = 32;                    // tile_dim1 = 32, tile_dim2 = 0 (2D)
  g1[5] = 128;                   // tensor_dim0_stride = 128 elements
  g1[6] = 0;
  g1[7] = 0;
  i32x4 zz = {0, 0, 0, 0};
#if TDM_ARGS6
  i32x8 z8 = {0, 0, 0, 0, 0, 0, 0, 0};
  __builtin_amdgcn_tensor_load_to_lds(g0, g1, zz, zz, z8, 0);
#else
  __builtin_amdgcn_tensor_load_to_lds(g0, g1, zz, zz, 0);
#endif
}
#endif

#if defined(USE_ASYNC)
DEVI void async_cp_b128(const _Float16* g, _Float16* l) {
  typedef __attribute__((address_space(1))) void gv_t;
  typedef __attribute__((address_space(3))) void lv_t;
  gv_t* gp = (gv_t*)(unsigned long long)(uintptr_t)g;
  lv_t* lp = (lv_t*)(unsigned)(uintptr_t)l;
  __builtin_amdgcn_global_load_async_to_lds_b128(gp, lp, 0, 0);
}
#endif

// ---------------- WMMA helpers (wave32, 16x16x32 f16 -> f32) ----------------
DEVI v8f wmma16(v16h a, v16h b, v8f c) {
  return __builtin_amdgcn_wmma_f32_16x16x32_f16(false, a, false, b, (short)0, c,
                                                false, false);
}

// A tile: 16(M) x 32(K), row-major source, p points at (m0,k0), leading dim ld.
// Lane L: M = L&15 ; K(e) = ((L>>4)<<4) + e
DEVI v16h load_a16(const _Float16* p, int ld) {
  int lane = threadIdx.x & 31;
  int m    = lane & 15;
  int khi  = (lane >> 4) << 4;
  const _Float16* r = p + (long)m * ld;
  v16h a;
#pragma unroll
  for (int e = 0; e < 16; ++e) a[e] = r[khi + e];
  return a;
}

// B tile: 32(K) x 16(N) taken as transpose of a 16(N) x 32(K) row-major block
// (i.e. B[k][n] = M2[n][k]).  Lane L: N = L&15 ; K(e) = ((L>>4)<<4) + e
DEVI v16h load_bt16(const _Float16* p, int ld) {
  int lane = threadIdx.x & 31;
  int n    = lane & 15;
  int kb   = (lane >> 4) << 4;
  const _Float16* r = p + (long)n * ld + kb;
  v16h b;
#pragma unroll
  for (int e = 0; e < 16; ++e) b[e] = r[e];
  return b;
}

// B tile: 32(K) x 16(N) row-major source (ld = row stride).
DEVI v16h load_b16(const _Float16* p, int ld) {
  int lane = threadIdx.x & 31;
  int n    = lane & 15;
  int kb   = (lane >> 4) << 4;
  v16h b;
#pragma unroll
  for (int e = 0; e < 16; ++e) b[e] = p[(long)(kb + e) * ld + n];
  return b;
}

// ---------------- utility kernels ----------------
__global__ void cvt_f16(const float* __restrict__ in, _Float16* __restrict__ out, int n) {
  int i = blockIdx.x * blockDim.x + threadIdx.x;
  if (i < n) out[i] = (_Float16)in[i];
}

// out[m][n] = sum_k A16[m][k] * W16[n][k] + bias[n]      (GEMM_NT, WMMA)
__global__ void __launch_bounds__(32)
gemm_nt_bias(const _Float16* __restrict__ A, const _Float16* __restrict__ W,
             const float* __restrict__ bias, float* __restrict__ D,
             int M, int Ncols, int K) {
  int ntiles = Ncols >> 4;
  int m0 = (blockIdx.x / ntiles) << 4;
  int n0 = (blockIdx.x % ntiles) << 4;
  v8f acc = {};
  for (int k = 0; k < K; k += 32) {
    __builtin_prefetch(A + (long)m0 * K + k + 64, 0, 1);
    v16h a = load_a16(A + (long)m0 * K + k, K);
    v16h b = load_bt16(W + (long)n0 * K + k, K);
    acc = wmma16(a, b, acc);
  }
  int lane = threadIdx.x & 31;
  int n = lane & 15, mb = (lane >> 4) << 3;
#pragma unroll
  for (int r = 0; r < 8; ++r)
    D[(long)(m0 + mb + r) * Ncols + n0 + n] = acc[r] + bias[n0 + n];
}

// RoPE on q/k, pack q,k -> (B,H,N,C) f16 ; v -> (B,N,H,C) f16
__global__ void rope_pack(const float* __restrict__ qraw, const float* __restrict__ kvraw,
                          _Float16* __restrict__ q16, _Float16* __restrict__ k16,
                          _Float16* __restrict__ v16) {
  int idx = blockIdx.x * blockDim.x + threadIdx.x;   // (b*N+n)*H*C + h*C + c
  int c  = idx & (Cc - 1);
  int h  = (idx >> 7) & (Hh - 1);
  int bn = idx >> 10;
  int n  = bn & (Nn - 1);
  int b  = bn >> 9;
  float ang  = (float)n * expf(-((float)(c & 63) / 64.0f) * LN1E4);
  float cv = cosf(ang), sv = sinf(ang);
  const float* qr = qraw + (long)bn * RQ + h * Cc;
  const float* kr = kvraw + (long)bn * RKV + h * 2 * Cc;
  float qv = qr[c];
  float qo = (c < 64) ? -qr[c + 64] : qr[c - 64];
  float kv = kr[c];
  float ko = (c < 64) ? -kr[c + 64] : kr[c - 64];
  float vv = kr[Cc + c];
  long qi = ((long)(b * Hh + h) * Nn + n) * Cc + c;
  q16[qi] = (_Float16)(qv * cv + qo * sv);
  k16[qi] = (_Float16)(kv * cv + ko * sv);
  v16[(long)bn * HC + h * Cc + c] = (_Float16)vv;
}

// q-points: rotate+translate, pack (B,H,N,32) f16 (cols 24..31 pre-zeroed), norms
__global__ void qpoints(const float* __restrict__ qpraw, const float* __restrict__ rot,
                        const float* __restrict__ trans, _Float16* __restrict__ qp16,
                        float* __restrict__ qn) {
  int idx = blockIdx.x * blockDim.x + threadIdx.x;   // (b*N+n)*64 + j
  int j = idx & 63;  int bn = idx >> 6;
  int h = j >> 3, p = j & 7;
  float vx = qpraw[(long)bn * RQP + j];
  float vy = qpraw[(long)bn * RQP + 64 + j];
  float vz = qpraw[(long)bn * RQP + 128 + j];
  const float* R = rot + (long)bn * 9;
  const float* t = trans + (long)bn * 3;
  float x = R[0]*vx + R[1]*vy + R[2]*vz + t[0];
  float y = R[3]*vx + R[4]*vy + R[5]*vz + t[1];
  float z = R[6]*vx + R[7]*vy + R[8]*vz + t[2];
  int b = bn >> 9, n = bn & (Nn - 1);
  _Float16* dst = qp16 + ((long)(b * Hh + h) * Nn + n) * 32 + p * 3;
  dst[0] = (_Float16)x; dst[1] = (_Float16)y; dst[2] = (_Float16)z;
  atomicAdd(&qn[(long)(b * Hh + h) * Nn + n], x*x + y*y + z*z);
}

// kv-points: split k_pts (pack like qp) and v_pts (B,N,H,48) f16
__global__ void kvpoints(const float* __restrict__ kvpraw, const float* __restrict__ rot,
                         const float* __restrict__ trans, _Float16* __restrict__ kp16,
                         float* __restrict__ kn, _Float16* __restrict__ vp16) {
  int idx = blockIdx.x * blockDim.x + threadIdx.x;   // (b*N+n)*160 + j
  int j = idx % 160;  int bn = idx / 160;
  int h = j / (PQKp + PVv);
  int pp = j % (PQKp + PVv);
  float vx = kvpraw[(long)bn * RKVP + j];
  float vy = kvpraw[(long)bn * RKVP + 160 + j];
  float vz = kvpraw[(long)bn * RKVP + 320 + j];
  const float* R = rot + (long)bn * 9;
  const float* t = trans + (long)bn * 3;
  float x = R[0]*vx + R[1]*vy + R[2]*vz + t[0];
  float y = R[3]*vx + R[4]*vy + R[5]*vz + t[1];
  float z = R[6]*vx + R[7]*vy + R[8]*vz + t[2];
  int b = bn >> 9, n = bn & (Nn - 1);
  if (pp < PQKp) {
    _Float16* dst = kp16 + ((long)(b * Hh + h) * Nn + n) * 32 + pp * 3;
    dst[0] = (_Float16)x; dst[1] = (_Float16)y; dst[2] = (_Float16)z;
    atomicAdd(&kn[(long)(b * Hh + h) * Nn + n], x*x + y*y + z*z);
  } else {
    int p = pp - PQKp;
    _Float16* dst = vp16 + ((long)bn * Hh + h) * 48 + p * 3;
    dst[0] = (_Float16)x; dst[1] = (_Float16)y; dst[2] = (_Float16)z;
  }
}

// bias[b,h,i,j] = z[b,i,j,:] . wb[h,:] + bb[h]   (single f32 pass over z)
// Also emits z16 (f16 copy of z) so the o_pair pass reads half the bytes and
// can stay resident in the 192 MB L2 (z16 = 128 MB).
__global__ void bias_z(const float* __restrict__ z, const float* __restrict__ wb,
                       const float* __restrict__ bb, float* __restrict__ biasT,
                       _Float16* __restrict__ z16) {
  long idx = (long)blockIdx.x * blockDim.x + threadIdx.x;  // (b*N+i)*N + j
  const float* zr = z + idx * CZc;
  _Float16* zw = z16 + idx * CZc;
  float acc[Hh] = {};
  for (int c = 0; c < CZc; ++c) {
    float zv = zr[c];
    zw[c] = (_Float16)zv;
#pragma unroll
    for (int h = 0; h < Hh; ++h) acc[h] += zv * wb[h * CZc + c];
  }
  int  j  = (int)(idx & (Nn - 1));
  long bi = idx >> 9;
  int  b  = (int)(bi >> 9);
  int  i  = (int)(bi & (Nn - 1));
#pragma unroll
  for (int h = 0; h < Hh; ++h)
    biasT[((long)(b * Hh + h) * Nn + i) * Nn + j] = acc[h] + bb[h];
}

// fused logits (q.k WMMA + point-dot WMMA + bias + mask) and softmax -> P16 f16
__global__ void __launch_bounds__(32)
attn_softmax(const _Float16* __restrict__ q16, const _Float16* __restrict__ k16,
             const _Float16* __restrict__ qp16, const _Float16* __restrict__ kp16,
             const float* __restrict__ qn, const float* __restrict__ kn,
             const float* __restrict__ biasT, const float* __restrict__ mask,
             const float* __restrict__ head_weights, _Float16* __restrict__ P16) {
  __shared__ float sl[16][Nn];                 // 32 KB logits tile
  int blk = blockIdx.x;
  int it  = blk & 31;
  int bh  = blk >> 5;                          // b*H + h
  int h   = bh & (Hh - 1), b = bh >> 3;
  int i0  = it << 4;
  int lane = threadIdx.x;
  float hw = logf(1.0f + expf(head_weights[h])) * SC_PT;

  const _Float16* qbase = q16 + ((long)bh * Nn + i0) * Cc;
  v16h aq0 = load_a16(qbase +  0, Cc);
  v16h aq1 = load_a16(qbase + 32, Cc);
  v16h aq2 = load_a16(qbase + 64, Cc);
  v16h aq3 = load_a16(qbase + 96, Cc);
  v16h ap  = load_a16(qp16 + ((long)bh * Nn + i0) * 32, 32);

  int n  = lane & 15;
  int mb = (lane >> 4) << 3;

  for (int jt = 0; jt < Nn / 16; ++jt) {
    int j0 = jt << 4;
    const _Float16* kbase = k16 + ((long)bh * Nn + j0) * Cc;
    v8f acc = {};
    acc = wmma16(aq0, load_bt16(kbase +  0, Cc), acc);
    acc = wmma16(aq1, load_bt16(kbase + 32, Cc), acc);
    acc = wmma16(aq2, load_bt16(kbase + 64, Cc), acc);
    acc = wmma16(aq3, load_bt16(kbase + 96, Cc), acc);
    v8f accp = {};
    accp = wmma16(ap, load_bt16(kp16 + ((long)bh * Nn + j0) * 32, 32), accp);

    int j   = j0 + n;
    float knj = kn[(long)bh * Nn + j];
    float mj  = mask[b * Nn + j];
#pragma unroll
    for (int r = 0; r < 8; ++r) {
      int m = mb + r;
      int i = i0 + m;
      float l = acc[r] * SC_QK
              + SC_B * biasT[((long)bh * Nn + i) * Nn + j]
              - 0.5f * hw * (qn[(long)bh * Nn + i] + knj - 2.0f * accp[r])
              + (mask[b * Nn + i] * mj - 1.0f) * 100000.0f;
      sl[m][j] = l;
    }
  }
  __syncthreads();
  if (lane < 16) {
    int i = i0 + lane;
    float mx = -3.0e38f;
    for (int j = 0; j < Nn; ++j) mx = fmaxf(mx, sl[lane][j]);
    float sum = 0.0f;
    for (int j = 0; j < Nn; ++j) {
      float e = expf(sl[lane][j] - mx);
      sl[lane][j] = e; sum += e;
    }
    float inv = 1.0f / sum;
    _Float16* pr = P16 + ((long)bh * Nn + i) * Nn;
    for (int j = 0; j < Nn; ++j) pr[j] = (_Float16)(sl[lane][j] * inv);
  }
}

// o = P @ v  -> feats[:, 0:HC] f16
__global__ void __launch_bounds__(32)
o_gemm(const _Float16* __restrict__ P16, const _Float16* __restrict__ v16,
       _Float16* __restrict__ feats) {
  int blk = blockIdx.x;                         // bh*(32*8) + it*8 + ct
  int ct = blk & 7;
  int it = (blk >> 3) & 31;
  int bh = blk >> 8;
  int h = bh & (Hh - 1), b = bh >> 3;
  int i0 = it << 4, c0 = ct << 4;
  const _Float16* Abase = P16 + ((long)bh * Nn + i0) * Nn;
  const _Float16* Bbase = v16 + (long)b * Nn * HC + h * Cc + c0;
  v8f acc = {};
  for (int k = 0; k < Nn; k += 32) {
    __builtin_prefetch(Abase + k + 64, 0, 1);
    v16h a  = load_a16(Abase + k, Nn);
    v16h bm = load_b16(Bbase + (long)k * HC, HC);
    acc = wmma16(a, bm, acc);
  }
  int lane = threadIdx.x & 31;
  int n = lane & 15, mb = (lane >> 4) << 3;
#pragma unroll
  for (int r = 0; r < 8; ++r)
    feats[(long)(b * Nn + i0 + mb + r) * FEAT + h * Cc + c0 + n] = (_Float16)acc[r];
}

// o_pt raw = P @ v_pts  (N=48 padded) -> optraw (B,H,N,48) f32
__global__ void __launch_bounds__(32)
opt_gemm(const _Float16* __restrict__ P16, const _Float16* __restrict__ vp16,
         float* __restrict__ optraw) {
  int blk = blockIdx.x;                         // bh*96 + it*3 + ct
  int ct = blk % 3;
  int it = (blk / 3) & 31;
  int bh = blk / 96;
  int h = bh & (Hh - 1), b = bh >> 3;
  int i0 = it << 4, c0 = ct << 4;
  const _Float16* Abase = P16 + ((long)bh * Nn + i0) * Nn;
  const _Float16* Bbase = vp16 + (long)b * Nn * Hh * 48 + h * 48 + c0;
  v8f acc = {};
  for (int k = 0; k < Nn; k += 32) {
    v16h a  = load_a16(Abase + k, Nn);
    v16h bm = load_b16(Bbase + (long)k * Hh * 48, Hh * 48);
    acc = wmma16(a, bm, acc);
  }
  int lane = threadIdx.x & 31;
  int n = lane & 15, mb = (lane >> 4) << 3;
#pragma unroll
  for (int r = 0; r < 8; ++r)
    optraw[((long)bh * Nn + i0 + mb + r) * 48 + c0 + n] = acc[r];
}

// inverse frame transform + norms -> feats[1024:1408]
__global__ void opt_final(const float* __restrict__ optraw, const float* __restrict__ rot,
                          const float* __restrict__ trans, _Float16* __restrict__ feats) {
  int idx = blockIdx.x * blockDim.x + threadIdx.x;  // (b*N+n)*H*PV + h*PV + p
  int p = idx % PVv;
  int t = idx / PVv;
  int h = t & (Hh - 1);
  int bn = t >> 3;
  int b = bn >> 9, n = bn & (Nn - 1);
  const float* src = optraw + ((long)(b * Hh + h) * Nn + n) * 48 + p * 3;
  const float* tr = trans + (long)bn * 3;
  float x = src[0] - tr[0], y = src[1] - tr[1], z = src[2] - tr[2];
  const float* R = rot + (long)bn * 9;
  float ox = R[0]*x + R[3]*y + R[6]*z;      // R^T
  float oy = R[1]*x + R[4]*y + R[7]*z;
  float oz = R[2]*x + R[5]*y + R[8]*z;
  float nr = sqrtf(ox*ox + oy*oy + oz*oz + 1e-8f);
  _Float16* f = feats + (long)bn * FEAT;
  f[HC +  0 + h * PVv + p] = (_Float16)ox;
  f[HC + 96 + h * PVv + p] = (_Float16)oy;
  f[HC + 192 + h * PVv + p] = (_Float16)oz;
  f[HC + 288 + h * PVv + p] = (_Float16)nr;
}

// o_pair: per (b,i): heads-as-M WMMA over z16[b,i,:,:].
// B-tiles (32 j-rows x 128 c, f16) staged into LDS by the Tensor Data Mover
// (double buffered), falling back to async-LDS loads, then to plain staging.
__global__ void __launch_bounds__(256)
opair_gemm(const _Float16* __restrict__ z16, const _Float16* __restrict__ P16,
           _Float16* __restrict__ feats) {
  __shared__ _Float16 zt[2][32][CZc];         // 2 x 8 KB ping-pong tiles
  int bi = blockIdx.x;                        // b*N + i
  int b = bi >> 9, i = bi & (Nn - 1);
  int tid = threadIdx.x;
  int lane = tid & 31;
  int c0 = (tid >> 5) << 4;                   // each of 8 waves owns 16 columns
  const _Float16* zbase = z16 + (long)bi * Nn * CZc;
  int m   = lane & 15;
  int khi = (lane >> 4) << 4;
  v8f acc = {};

#if defined(USE_TDM)
  if (tid < 32)
    tdm_load_tile_32x128h((unsigned)(uintptr_t)&zt[0][0][0],
                          (unsigned long long)(uintptr_t)zbase);
  int cur = 0;
#endif
  for (int k0 = 0; k0 < Nn; k0 += 32) {
#if defined(USE_TDM)
    if (tid < 32) wait_tensor0();
    __syncthreads();                          // tile[cur] ready for all waves
    if (k0 + 32 < Nn && tid < 32)
      tdm_load_tile_32x128h((unsigned)(uintptr_t)&zt[cur ^ 1][0][0],
                            (unsigned long long)(uintptr_t)(zbase + (long)(k0 + 32) * CZc));
    const _Float16* ztc = &zt[cur][0][0];
#elif defined(USE_ASYNC)
    __syncthreads();
    // 8 KB tile = 512 x b128 lane transfers; 2 per thread
#pragma unroll
    for (int it2 = 0; it2 < 2; ++it2) {
      int e = (it2 << 8) + tid;               // 16-byte chunk index
      int jr = e >> 4, cc = (e & 15) << 3;    // 8 halves per chunk
      async_cp_b128(zbase + (long)(k0 + jr) * CZc + cc, &zt[0][jr][cc]);
    }
    wait_async0();
    __syncthreads();
    const _Float16* ztc = &zt[0][0][0];
#else
    __syncthreads();
    for (int e = tid; e < 32 * CZc; e += 256) {
      int jr = e >> 7, cc = e & (CZc - 1);
      zt[0][jr][cc] = zbase[(long)(k0 + jr) * CZc + cc];
    }
    __syncthreads();
    const _Float16* ztc = &zt[0][0][0];
#endif
    v16h a;
#pragma unroll
    for (int e = 0; e < 16; ++e) {
      int kk = khi + e;
      a[e] = (m < Hh) ? P16[((long)(b * Hh + m) * Nn + i) * Nn + k0 + kk]
                      : (_Float16)0.0f;
    }
    v16h bm = load_b16(ztc + c0, CZc);
    acc = wmma16(a, bm, acc);
#if defined(USE_TDM)
    __syncthreads();                          // reads done before buffer reuse
    cur ^= 1;
#endif
  }
  int n = lane & 15, mb = (lane >> 4) << 3;
#pragma unroll
  for (int r = 0; r < 8; ++r) {
    int hrow = mb + r;
    if (hrow < Hh)
      feats[(long)bi * FEAT + (HC + 384) + hrow * CZc + c0 + n] = (_Float16)acc[r];
  }
}

// final projection: out = feats16 @ wout16^T + bout
__global__ void __launch_bounds__(32)
out_gemm(const _Float16* __restrict__ feats, const _Float16* __restrict__ wout16,
         const float* __restrict__ bout, float* __restrict__ out) {
  int blk = blockIdx.x;
  int nt = blk & 15, mt = blk >> 4;
  int m0 = mt << 4, n0 = nt << 4;
  v8f acc = {};
  for (int k = 0; k < FEAT; k += 32) {
    __builtin_prefetch(feats + (long)m0 * FEAT + k + 64, 0, 1);
    v16h a  = load_a16(feats + (long)m0 * FEAT + k, FEAT);
    v16h bm = load_bt16(wout16 + (long)n0 * FEAT + k, FEAT);
    acc = wmma16(a, bm, acc);
  }
  int lane = threadIdx.x & 31;
  int n = lane & 15, mb = (lane >> 4) << 3;
#pragma unroll
  for (int r = 0; r < 8; ++r)
    out[(long)(m0 + mb + r) * CSc + n0 + n] = acc[r] + bout[n0 + n];
}

// ---------------- launcher ----------------
extern "C" void kernel_launch(void* const* d_in, const int* in_sizes, int n_in,
                              void* d_out, int out_size, void* d_ws, size_t ws_size,
                              hipStream_t stream) {
  const float* s     = (const float*)d_in[0];
  const float* z     = (const float*)d_in[1];
  const float* rot   = (const float*)d_in[2];
  const float* trans = (const float*)d_in[3];
  const float* mask  = (const float*)d_in[4];
  const float* wq    = (const float*)d_in[5];
  const float* bq    = (const float*)d_in[6];
  const float* wkv   = (const float*)d_in[7];
  const float* bkv   = (const float*)d_in[8];
  const float* wqp   = (const float*)d_in[9];
  const float* bqp   = (const float*)d_in[10];
  const float* wkvp  = (const float*)d_in[11];
  const float* bkvp  = (const float*)d_in[12];
  const float* wb    = (const float*)d_in[13];
  const float* bb    = (const float*)d_in[14];
  const float* hwts  = (const float*)d_in[15];
  const float* wout  = (const float*)d_in[16];
  const float* bout  = (const float*)d_in[17];
  float* out = (float*)d_out;

  // workspace carve-up (~186 MB total; z16 is the 128 MB tail)
  char* w = (char*)d_ws;
  _Float16* s16    = (_Float16*)w; w += (size_t)2 * Bb * Nn * CSc;
  _Float16* wq16   = (_Float16*)w; w += (size_t)2 * RQ * CSc;
  _Float16* wkv16  = (_Float16*)w; w += (size_t)2 * RKV * CSc;
  _Float16* wqp16  = (_Float16*)w; w += (size_t)2 * RQP * CSc;
  _Float16* wkvp16 = (_Float16*)w; w += (size_t)2 * RKVP * CSc;
  _Float16* wout16 = (_Float16*)w; w += (size_t)2 * CSc * FEAT;
  float* qraw   = (float*)w; w += (size_t)4 * Bb * Nn * RQ;
  float* kvraw  = (float*)w; w += (size_t)4 * Bb * Nn * RKV;
  float* qpraw  = (float*)w; w += (size_t)4 * Bb * Nn * RQP;
  float* kvpraw = (float*)w; w += (size_t)4 * Bb * Nn * RKVP;
  _Float16* q16  = (_Float16*)w; w += (size_t)2 * Bb * Hh * Nn * Cc;
  _Float16* k16  = (_Float16*)w; w += (size_t)2 * Bb * Hh * Nn * Cc;
  _Float16* v16  = (_Float16*)w; w += (size_t)2 * Bb * Nn * HC;
  _Float16* qp16 = (_Float16*)w; w += (size_t)2 * Bb * Hh * Nn * 32;
  _Float16* kp16 = (_Float16*)w; w += (size_t)2 * Bb * Hh * Nn * 32;
  _Float16* vp16 = (_Float16*)w; w += (size_t)2 * Bb * Nn * Hh * 48;
  float* qn = (float*)w; w += (size_t)4 * Bb * Hh * Nn;
  float* kn = (float*)w; w += (size_t)4 * Bb * Hh * Nn;
  float* biasT = (float*)w; w += (size_t)4 * Bb * Hh * Nn * Nn;
  _Float16* P16 = (_Float16*)w; w += (size_t)2 * Bb * Hh * Nn * Nn;
  float* optraw = (float*)w; w += (size_t)4 * Bb * Hh * Nn * 48;
  _Float16* feats = (_Float16*)w; w += (size_t)2 * Bb * Nn * FEAT;
  _Float16* z16   = (_Float16*)w; w += (size_t)2 * Bb * Nn * Nn * CZc;

  // zero pad-sensitive buffers (cols 24..31 / 36..47 of point tiles, norm accum)
  hipMemsetAsync(qp16, 0, (size_t)2 * Bb * Hh * Nn * 32, stream);
  hipMemsetAsync(kp16, 0, (size_t)2 * Bb * Hh * Nn * 32, stream);
  hipMemsetAsync(vp16, 0, (size_t)2 * Bb * Nn * Hh * 48, stream);
  hipMemsetAsync(qn, 0, (size_t)4 * Bb * Hh * Nn, stream);
  hipMemsetAsync(kn, 0, (size_t)4 * Bb * Hh * Nn, stream);

  auto cvt = [&](const float* src, _Float16* dst, int n) {
    cvt_f16<<<(n + 255) / 256, 256, 0, stream>>>(src, dst, n);
  };
  cvt(s, s16, Bb * Nn * CSc);
  cvt(wq, wq16, RQ * CSc);
  cvt(wkv, wkv16, RKV * CSc);
  cvt(wqp, wqp16, RQP * CSc);
  cvt(wkvp, wkvp16, RKVP * CSc);
  cvt(wout, wout16, CSc * FEAT);

  const int M = Bb * Nn;   // 1024
  gemm_nt_bias<<<(M / 16) * (RQ / 16),   32, 0, stream>>>(s16, wq16,   bq,   qraw,   M, RQ,   CSc);
  gemm_nt_bias<<<(M / 16) * (RKV / 16),  32, 0, stream>>>(s16, wkv16,  bkv,  kvraw,  M, RKV,  CSc);
  gemm_nt_bias<<<(M / 16) * (RQP / 16),  32, 0, stream>>>(s16, wqp16,  bqp,  qpraw,  M, RQP,  CSc);
  gemm_nt_bias<<<(M / 16) * (RKVP / 16), 32, 0, stream>>>(s16, wkvp16, bkvp, kvpraw, M, RKVP, CSc);

  rope_pack<<<(Bb * Nn * HC) / 256, 256, 0, stream>>>(qraw, kvraw, q16, k16, v16);
  qpoints<<<(Bb * Nn * 64) / 256, 256, 0, stream>>>(qpraw, rot, trans, qp16, qn);
  kvpoints<<<(Bb * Nn * 160) / 256, 256, 0, stream>>>(kvpraw, rot, trans, kp16, kn, vp16);

  bias_z<<<(Bb * Nn * Nn) / 256, 256, 0, stream>>>(z, wb, bb, biasT, z16);

  attn_softmax<<<Bb * Hh * (Nn / 16), 32, 0, stream>>>(q16, k16, qp16, kp16, qn, kn,
                                                       biasT, mask, hwts, P16);

  o_gemm<<<Bb * Hh * (Nn / 16) * (Cc / 16), 32, 0, stream>>>(P16, v16, feats);
  opt_gemm<<<Bb * Hh * (Nn / 16) * 3, 32, 0, stream>>>(P16, vp16, optraw);
  opt_final<<<(Bb * Nn * Hh * PVv) / 256, 256, 0, stream>>>(optraw, rot, trans, feats);
  opair_gemm<<<Bb * Nn, 256, 0, stream>>>(z16, P16, feats);

  out_gemm<<<(M / 16) * (CSc / 16), 32, 0, stream>>>(feats, wout16, bout, out);
}
